// OmniAttention_64768106823999
// MI455X (gfx1250) — matrix operand order repaired and checked by reading, not compiled, not run on hardware
//
#include <hip/hip_runtime.h>
#include <hip/hip_bf16.h>

typedef _Float16 hf;
typedef _Float16 h4  __attribute__((ext_vector_type(4)));
typedef _Float16 h8  __attribute__((ext_vector_type(8)));
typedef _Float16 v16h __attribute__((ext_vector_type(16)));
typedef float    v8f __attribute__((ext_vector_type(8)));
typedef float    f4  __attribute__((ext_vector_type(4)));

#define BATCH 4
#define SEQ   1024
#define EMB   1280
#define NH    16
#define HD    80
#define HDP   96   // head dim padded to 3 x 32 for WMMA K-steps

union V16u { v16h v; h8 h[2]; };
__device__ __forceinline__ v16h make_v16(h8 lo, h8 hi) { V16u u; u.h[0] = lo; u.h[1] = hi; return u.v; }
__device__ __forceinline__ v8f vzero8() { v8f z; for (int i = 0; i < 8; ++i) z[i] = 0.f; return z; }

// Generic LDS pointer -> 32-bit LDS byte offset (ISA 10.2: low 32 bits of the address).
__device__ __forceinline__ unsigned lds_off32(const void* p) { return (unsigned)(size_t)p; }

// Async copy 16B/lane global -> LDS (gfx1250), tracked by ASYNCcnt.
__device__ __forceinline__ void async_g2l_b128(unsigned lds_addr, const void* gaddr) {
    asm volatile("global_load_async_to_lds_b128 %0, %1, off"
                 :: "v"(lds_addr), "v"(gaddr) : "memory");
}
__device__ __forceinline__ void wait_asynccnt0() {
    asm volatile("s_wait_asynccnt 0x0" ::: "memory");
}

// ---------------------------------------------------------------------------
// One-shot fp32 -> f16 conversion (vectorized x4).
// ---------------------------------------------------------------------------
__global__ void cvt_f32_f16(const float* __restrict__ src, hf* __restrict__ dst, int n)
{
    int i = (blockIdx.x * blockDim.x + threadIdx.x) * 4;
    if (i >= n) return;
    f4 v = *(const f4*)&src[i];
    h4 o;
    for (int e = 0; e < 4; ++e) o[e] = (hf)v[e];
    *(h4*)&dst[i] = o;
}

// ---------------------------------------------------------------------------
// GEMM: C[M,N] = A[M,K] (f16) @ W[N,K]^T (f16) + bias[N], f16 WMMA, f32 accum.
// Block = 256 threads = 8 waves; block tile 128x128; wave tile 32x64 = 2x4 WMMA.
// K step 32. Staging: double-buffered global_load_async_to_lds_b128 overlapped
// with WMMA compute; fragments built from ds_load_b128.
// ---------------------------------------------------------------------------
__global__ void __launch_bounds__(256)
gemm_bias_wmma(const hf* __restrict__ A, const hf* __restrict__ W,
               const float* __restrict__ bias, float* __restrict__ C,
               int M, int N, int K)
{
    const int LDT = 40; // lds row stride in halves (80B rows; 16B-aligned b128 slots)
    __shared__ hf Ash[2][128 * 40];
    __shared__ hf Bsh[2][128 * 40];

    const int tid  = threadIdx.x;
    const int lane = tid & 31;
    const int w    = tid >> 5;
    const int hi   = lane >> 4;
    const int ln   = lane & 15;
    const int wr   = w >> 1;   // 0..3 -> row offset wr*32
    const int wc   = w & 1;    // 0..1 -> col offset wc*64
    const int m0   = blockIdx.x * 128;
    const int n0   = blockIdx.y * 128;

    // Stage one 128x32-half tile pair into LDS buffer `buf` (async, no wait).
    auto stage = [&](int k0, int buf) {
        for (int it = 0; it < 2; ++it) {
            int idx = tid + it * 256;          // 512 h8 slots per tile
            int r   = idx >> 2;
            int c8  = (idx & 3) * 8;
            async_g2l_b128(lds_off32(&Ash[buf][r * LDT + c8]),
                           &A[(size_t)(m0 + r) * K + (k0 + c8)]);
            async_g2l_b128(lds_off32(&Bsh[buf][r * LDT + c8]),
                           &W[(size_t)(n0 + r) * K + (k0 + c8)]);
        }
    };

    v8f acc[2][4];
    for (int i = 0; i < 2; ++i)
        for (int j = 0; j < 4; ++j)
            acc[i][j] = vzero8();

    stage(0, 0);
    wait_asynccnt0();
    __syncthreads();

    int buf = 0;
    for (int k0 = 0; k0 < K; k0 += 32, buf ^= 1) {
        if (k0 + 32 < K) stage(k0 + 32, buf ^ 1);   // overlap next fill with compute

        v16h af[2], bf[4];
        for (int i = 0; i < 2; ++i) {
            const hf* pa = &Ash[buf][(wr * 32 + i * 16 + ln) * LDT + hi * 8];
            af[i] = make_v16(*(const h8*)pa, *(const h8*)(pa + 16));
        }
        for (int j = 0; j < 4; ++j) {
            const hf* pb = &Bsh[buf][(wc * 64 + j * 16 + ln) * LDT + hi * 8];
            bf[j] = make_v16(*(const h8*)pb, *(const h8*)(pb + 16));
        }
        for (int i = 0; i < 2; ++i)
            for (int j = 0; j < 4; ++j)
                acc[i][j] = __builtin_amdgcn_wmma_f32_16x16x32_f16(
                    false, af[i], false, bf[j], (short)0, acc[i][j], false, false);

        wait_asynccnt0();   // next buffer landed
        __syncthreads();    // everyone done reading current buffer
    }

    for (int i = 0; i < 2; ++i)
        for (int j = 0; j < 4; ++j) {
            int n  = n0 + wc * 64 + j * 16 + ln;
            float bv = bias[n];
            for (int r = 0; r < 8; ++r) {
                int m = m0 + wr * 32 + i * 16 + r + hi * 8;
                C[(size_t)m * N + n] = acc[i][j][r] + bv;
            }
        }
}

// ---------------------------------------------------------------------------
// RoPE + relayout: raw fp32 q/k/v [B*S, E] ->
//   Qh,Kh: f16 [B,H,S,HDP] (zero-padded d>=80), VT: f16 [B,H,HD,S] (V transposed)
// ---------------------------------------------------------------------------
__global__ void rope_relayout(const float* __restrict__ qraw, const float* __restrict__ kraw,
                              const float* __restrict__ vraw, const float* __restrict__ cosT,
                              const float* __restrict__ sinT, hf* __restrict__ Qh,
                              hf* __restrict__ Kh, hf* __restrict__ VT)
{
    int idx = blockIdx.x * blockDim.x + threadIdx.x;
    const int total = BATCH * NH * SEQ * HDP;
    if (idx >= total) return;
    int d = idx % HDP;
    int s = (idx / HDP) % SEQ;
    int h = (idx / (HDP * SEQ)) % NH;
    int b = idx / (HDP * SEQ * NH);

    size_t qk_off = ((size_t)(b * NH + h) * SEQ + s) * HDP + d;
    if (d >= HD) { Qh[qk_off] = (hf)0.f; Kh[qk_off] = (hf)0.f; return; }

    size_t base = ((size_t)b * SEQ + s) * EMB + h * HD;
    float q = qraw[base + d];
    float k = kraw[base + d];
    float v = vraw[base + d];
    float c  = cosT[s * HD + d];
    float sn = sinT[s * HD + d];
    float qr = (d < HD / 2) ? -qraw[base + d + HD / 2] : qraw[base + d - HD / 2];
    float kr = (d < HD / 2) ? -kraw[base + d + HD / 2] : kraw[base + d - HD / 2];
    Qh[qk_off] = (hf)(q * c + qr * sn);
    Kh[qk_off] = (hf)(k * c + kr * sn);
    VT[((size_t)(b * NH + h) * HD + d) * SEQ + s] = (hf)v;
}

// ---------------------------------------------------------------------------
// Flash attention: block = 128 threads = 4 waves; block handles (b, h, 64 q rows);
// each wave owns 16 q rows. Key blocks of 64. Online softmax in C-fragment layout.
// QK^T: 12 WMMA / key block; PV: 10 WMMA / key block (P via per-wave LDS transpose).
// Output written directly in f16 for the output-projection GEMM.
// ---------------------------------------------------------------------------
__global__ void __launch_bounds__(128)
flash_attn_wmma(const hf* __restrict__ Qh, const hf* __restrict__ Kh,
                const hf* __restrict__ VT, const int* __restrict__ cu,
                hf* __restrict__ Obuf)
{
    __shared__ hf Pl[4 * 16 * 64]; // per-wave 16x64 probability tiles

    const int tid  = threadIdx.x;
    const int lane = tid & 31;
    const int w    = tid >> 5;
    const int hi   = lane >> 4;
    const int ln   = lane & 15;
    const int h    = blockIdx.y;
    const int b    = blockIdx.z;
    const int len  = cu[b + 1] - cu[b];
    const int q0   = blockIdx.x * 64 + w * 16;
    const float scale = 0.1118033988749895f; // 1/sqrt(80)

    const hf* Qb = Qh + (size_t)(b * NH + h) * SEQ * HDP;
    const hf* Kb = Kh + (size_t)(b * NH + h) * SEQ * HDP;
    const hf* Vb = VT + (size_t)(b * NH + h) * HD * SEQ;

    // Q fragments (kept in registers for the whole kernel)
    v16h aq[3];
    {
        const hf* qrow = Qb + (size_t)(q0 + ln) * HDP;
        for (int ks = 0; ks < 3; ++ks) {
            const hf* p = qrow + ks * 32 + hi * 8;
            aq[ks] = make_v16(*(const h8*)p, *(const h8*)(p + 16));
        }
    }

    float m_run[8], l_run[8];
    v8f accO[5];
    for (int r = 0; r < 8; ++r) { m_run[r] = -__builtin_huge_valf(); l_run[r] = 0.f; }
    for (int t = 0; t < 5; ++t) accO[t] = vzero8();

    for (int jb = 0; jb < SEQ; jb += 64) {
        // ---- scores tile: 16 x 64 ----
        v8f accS[4];
        for (int t = 0; t < 4; ++t) accS[t] = vzero8();
        for (int ks = 0; ks < 3; ++ks) {
            for (int tj = 0; tj < 4; ++tj) {
                const hf* p = Kb + (size_t)(jb + tj * 16 + ln) * HDP + ks * 32 + hi * 8;
                v16h bk = make_v16(*(const h8*)p, *(const h8*)(p + 16));
                accS[tj] = __builtin_amdgcn_wmma_f32_16x16x32_f16(
                    false, aq[ks], false, bk, (short)0, accS[tj], false, false);
            }
        }

        // ---- scale + varlen mask + row max ----
        float mnew[8];
        for (int r = 0; r < 8; ++r) mnew[r] = -__builtin_huge_valf();
        for (int tj = 0; tj < 4; ++tj) {
            int key = jb + tj * 16 + ln;
            bool kv = key < len;
            for (int r = 0; r < 8; ++r) {
                bool qv = (q0 + r + hi * 8) < len;
                float s = (kv && qv) ? accS[tj][r] * scale : -1.0e9f;
                accS[tj][r] = s;
                mnew[r] = fmaxf(mnew[r], s);
            }
        }
        for (int off = 1; off < 16; off <<= 1)
            for (int r = 0; r < 8; ++r)
                mnew[r] = fmaxf(mnew[r], __shfl_xor(mnew[r], off, 32));

        // ---- online softmax update ----
        float sf[8], lad[8];
        for (int r = 0; r < 8; ++r) {
            float m2 = fmaxf(m_run[r], mnew[r]);
            sf[r] = __expf(m_run[r] - m2);
            m_run[r] = m2;
            lad[r] = 0.f;
        }
        for (int tj = 0; tj < 4; ++tj)
            for (int r = 0; r < 8; ++r) {
                float p = __expf(accS[tj][r] - m_run[r]);
                accS[tj][r] = p;
                lad[r] += p;
            }
        for (int off = 1; off < 16; off <<= 1)
            for (int r = 0; r < 8; ++r)
                lad[r] += __shfl_xor(lad[r], off, 32);
        for (int r = 0; r < 8; ++r) l_run[r] = l_run[r] * sf[r] + lad[r];
        for (int t = 0; t < 5; ++t)
            for (int r = 0; r < 8; ++r) accO[t][r] *= sf[r];

        // ---- P (C-layout) -> LDS -> A-layout fragments ----
        hf* pw = &Pl[w * 1024];
        for (int tj = 0; tj < 4; ++tj)
            for (int r = 0; r < 8; ++r)
                pw[(r + hi * 8) * 64 + tj * 16 + ln] = (hf)accS[tj][r];
        // same-wave LDS ops are in order: no barrier needed

        for (int kb = 0; kb < 2; ++kb) {
            const hf* pr = &Pl[w * 1024 + ln * 64 + kb * 32 + hi * 8];
            v16h ap = make_v16(*(const h8*)pr, *(const h8*)(pr + 16));
            for (int tv = 0; tv < 5; ++tv) {
                const hf* vp = Vb + (size_t)(tv * 16 + ln) * SEQ + jb + kb * 32 + hi * 8;
                v16h bv = make_v16(*(const h8*)vp, *(const h8*)(vp + 16));
                accO[tv] = __builtin_amdgcn_wmma_f32_16x16x32_f16(
                    false, ap, false, bv, (short)0, accO[tv], false, false);
            }
        }
    }

    // ---- normalize + write f16 [B,S,H*D] ----
    for (int r = 0; r < 8; ++r) {
        float inv = 1.f / l_run[r];
        int m = q0 + r + hi * 8;
        for (int tv = 0; tv < 5; ++tv) {
            int d = tv * 16 + ln;
            Obuf[((size_t)b * SEQ + m) * EMB + h * HD + d] = (hf)(accO[tv][r] * inv);
        }
    }
}

// ---------------------------------------------------------------------------
extern "C" void kernel_launch(void* const* d_in, const int* in_sizes, int n_in,
                              void* d_out, int out_size, void* d_ws, size_t ws_size,
                              hipStream_t stream)
{
    const float* x    = (const float*)d_in[0];
    const int*   cu   = (const int*)d_in[1];
    const float* cosT = (const float*)d_in[2];
    const float* sinT = (const float*)d_in[3];
    const float* wq   = (const float*)d_in[4];
    const float* bq   = (const float*)d_in[5];
    const float* wk   = (const float*)d_in[6];
    const float* bk   = (const float*)d_in[7];
    const float* wv   = (const float*)d_in[8];
    const float* bv   = (const float*)d_in[9];
    const float* wo   = (const float*)d_in[10];
    const float* bo   = (const float*)d_in[11];
    float* out = (float*)d_out;

    const size_t M  = (size_t)BATCH * SEQ;   // 4096
    const size_t EE = (size_t)EMB * EMB;     // 1280*1280
    char* ws = (char*)d_ws;
    float* qraw = (float*)ws; ws += M * EMB * sizeof(float);
    float* kraw = (float*)ws; ws += M * EMB * sizeof(float);
    float* vraw = (float*)ws; ws += M * EMB * sizeof(float);
    hf* xh  = (hf*)ws; ws += M * EMB * sizeof(hf);
    hf* wqh = (hf*)ws; ws += EE * sizeof(hf);
    hf* wkh = (hf*)ws; ws += EE * sizeof(hf);
    hf* wvh = (hf*)ws; ws += EE * sizeof(hf);
    hf* woh = (hf*)ws; ws += EE * sizeof(hf);
    hf* Qh  = (hf*)ws; ws += (size_t)BATCH * NH * SEQ * HDP * sizeof(hf);
    hf* Kh  = (hf*)ws; ws += (size_t)BATCH * NH * SEQ * HDP * sizeof(hf);
    hf* VT  = (hf*)ws; ws += (size_t)BATCH * NH * HD * SEQ * sizeof(hf);
    hf* attn = (hf*)ws;

    // One-shot f16 conversion of GEMM operands.
    int nx = (int)(M * EMB), nw = (int)EE;
    cvt_f32_f16<<<(nx / 4 + 255) / 256, 256, 0, stream>>>(x,  xh,  nx);
    cvt_f32_f16<<<(nw / 4 + 255) / 256, 256, 0, stream>>>(wq, wqh, nw);
    cvt_f32_f16<<<(nw / 4 + 255) / 256, 256, 0, stream>>>(wk, wkh, nw);
    cvt_f32_f16<<<(nw / 4 + 255) / 256, 256, 0, stream>>>(wv, wvh, nw);
    cvt_f32_f16<<<(nw / 4 + 255) / 256, 256, 0, stream>>>(wo, woh, nw);

    dim3 gg(M / 128, EMB / 128);
    gemm_bias_wmma<<<gg, 256, 0, stream>>>(xh, wqh, bq, qraw, (int)M, EMB, EMB);
    gemm_bias_wmma<<<gg, 256, 0, stream>>>(xh, wkh, bk, kraw, (int)M, EMB, EMB);
    gemm_bias_wmma<<<gg, 256, 0, stream>>>(xh, wvh, bv, vraw, (int)M, EMB, EMB);

    int tot = BATCH * NH * SEQ * HDP;
    rope_relayout<<<(tot + 255) / 256, 256, 0, stream>>>(qraw, kraw, vraw, cosT, sinT, Qh, Kh, VT);

    dim3 ga(SEQ / 64, NH, BATCH);
    flash_attn_wmma<<<ga, 128, 0, stream>>>(Qh, Kh, VT, cu, attn);

    // Output projection reuses the f16 GEMM on the f16 attention output.
    gemm_bias_wmma<<<gg, 256, 0, stream>>>(attn, woh, bo, out, (int)M, EMB, EMB);
}
